// BertLayer_65231963292395
// MI455X (gfx1250) — matrix operand order repaired and checked by reading, not compiled, not run on hardware
//
#include <hip/hip_runtime.h>

#define B_   2
#define S_   2048
#define H_   768
#define NH_  12
#define DH_  64
#define FF_  3072
#define RB_  32
#define MTOK (B_ * S_)   // 4096 tokens

typedef __bf16 v16bf __attribute__((ext_vector_type(16)));
typedef __bf16 v8bf  __attribute__((ext_vector_type(8)));
typedef float  v8f   __attribute__((ext_vector_type(8)));
typedef unsigned int u32x4 __attribute__((ext_vector_type(4)));
typedef int          i32x8 __attribute__((ext_vector_type(8)));
typedef int          i32x4 __attribute__((ext_vector_type(4)));

// LDS (addrspace 3) flavors: 32-bit pointers, loads lower to ds_load_*.
typedef __attribute__((address_space(3))) __bf16 lds_bf16;
typedef __attribute__((address_space(3))) v8bf   lds_v8bf;

// ---------------------------------------------------------------- WMMA core
__device__ __forceinline__ v8f wmma_bf16(v16bf a, v16bf b, v8f c) {
  return __builtin_amdgcn_wmma_f32_16x16x32_bf16(
      /*neg_a=*/false, a, /*neg_b=*/false, b,
      /*c_mod=*/(short)0, c, /*reuse_a=*/false, /*reuse_b=*/false);
}

// A fragment (16x32 bf16): lane row = lane&15, half = lane>>4.
__device__ __forceinline__ v16bf load_a_bf16(const __bf16* row, int k0, int half) {
  v8bf s0 = *(const v8bf*)(row + k0 + half * 8);
  v8bf s1 = *(const v8bf*)(row + k0 + 16 + half * 8);
  v16bf a;
#pragma unroll
  for (int i = 0; i < 8; ++i) { a[i] = s0[i]; a[i + 8] = s1[i]; }
  return a;
}

// B fragment (32x16 bf16): lane col = lane&15; K = k0 + 8*half + e (e<8), k0 + 16 + 8*half + (e-8) (e>=8)
__device__ __forceinline__ v16bf load_b_bf16(const __bf16* row, int k0, int half) {
  v8bf s0 = *(const v8bf*)(row + k0 + half * 8);
  v8bf s1 = *(const v8bf*)(row + k0 + 16 + half * 8);
  v16bf b;
#pragma unroll
  for (int i = 0; i < 8; ++i) { b[i] = s0[i]; b[i + 8] = s1[i]; }
  return b;
}

// B fragment sourced from LDS (addrspace 3) -> ds_load_b128 x2
__device__ __forceinline__ v16bf load_b_lds(const lds_bf16* row, int k0, int half) {
  const lds_v8bf* p = (const lds_v8bf*)(row + k0 + half * 8);
  const lds_v8bf* q = (const lds_v8bf*)(row + k0 + 16 + half * 8);
  v8bf s0 = p[0];
  v8bf s1 = q[0];
  v16bf b;
#pragma unroll
  for (int i = 0; i < 8; ++i) { b[i] = s0[i]; b[i + 8] = s1[i]; }
  return b;
}

// A fragment sourced from f32 data, converted to bf16 in registers.
__device__ __forceinline__ v16bf load_a_f32(const float* row, int k0, int half) {
  const float* p0 = row + k0 + half * 8;
  const float* p1 = row + k0 + 16 + half * 8;
  v16bf a;
#pragma unroll
  for (int i = 0; i < 8; ++i) { a[i] = (__bf16)p0[i]; a[i + 8] = (__bf16)p1[i]; }
  return a;
}

// ---------------------------------------------------------------- reductions
__device__ __forceinline__ float wave_reduce_sum(float x) {
#pragma unroll
  for (int o = 16; o > 0; o >>= 1) x += __shfl_xor(x, o, 32);
  return x;
}
__device__ __forceinline__ float wave_reduce_max(float x) {
#pragma unroll
  for (int o = 16; o > 0; o >>= 1) x = fmaxf(x, __shfl_xor(x, o, 32));
  return x;
}

// ---------------------------------------------------------------- converters
__global__ void k_cvt(const float* __restrict__ s, __bf16* __restrict__ d, int n) {
  int i = blockIdx.x * 256 + threadIdx.x;
  if (i < n) d[i] = (__bf16)s[i];
}
__global__ void k_cvt_pad(const float* __restrict__ s, __bf16* __restrict__ d,
                          int rows, int cols, int padrows) {
  int i = blockIdx.x * 256 + threadIdx.x;
  if (i >= padrows * cols) return;
  int r = i / cols;
  d[i] = (r < rows) ? (__bf16)s[i] : (__bf16)0.0f;
}

// ---------------------------------------------------------------- QKV
__global__ void k_qkv(const __bf16* __restrict__ hb,
                      const __bf16* __restrict__ Wq, const __bf16* __restrict__ Wk,
                      const __bf16* __restrict__ Wv,
                      const float* __restrict__ bq, const float* __restrict__ bv,
                      __bf16* __restrict__ qb, __bf16* __restrict__ kb,
                      __bf16* __restrict__ vb) {
  int wave = blockIdx.x * 8 + (threadIdx.x >> 5);
  int lane = threadIdx.x & 31;
  int half = lane >> 4, nn = lane & 15;
  const int NT = H_ / 16;               // 48
  const int TILES = (MTOK / 16) * NT;   // 12288
  int which = wave / TILES;             // 0=q 1=k 2=v (wave-uniform)
  int t = wave % TILES;
  int tm = (t / NT) * 16, tn = (t % NT) * 16;
  int n = tn + nn;
  const __bf16* W = (which == 0) ? Wq : (which == 1) ? Wk : Wv;
  const __bf16* wrow = W + (size_t)n * H_;
  const __bf16* arow = hb + (size_t)(tm + nn) * H_;
  v8f acc = {};
#pragma unroll 4
  for (int k0 = 0; k0 < H_; k0 += 32)
    acc = wmma_bf16(load_a_bf16(arow, k0, half), load_b_bf16(wrow, k0, half), acc);
  int head = n >> 6, d = n & 63;
#pragma unroll
  for (int r = 0; r < 8; ++r) {
    int m = tm + r + 8 * half;
    int b = m >> 11, s = m & (S_ - 1);
    float c = acc[r];
    if (which == 0)
      qb[((size_t)(b * NH_ + head) * S_ + s) * DH_ + d] = (__bf16)((c + bq[n]) * 0.125f);
    else if (which == 1)
      kb[((size_t)(b * NH_ + head) * S_ + s) * DH_ + d] = (__bf16)c;
    else
      vb[((size_t)(b * NH_ + head) * DH_ + d) * S_ + s] = (__bf16)(c + bv[n]);
  }
}

// -------------------------------------------------- rel-pos bias: [S*S,32]x[32,16pad]
__global__ void k_relpos(const float* __restrict__ rel, const __bf16* __restrict__ rpw,
                         __bf16* __restrict__ bias) {
  int wave = blockIdx.x * 8 + (threadIdx.x >> 5);
  int lane = threadIdx.x & 31;
  int half = lane >> 4, nn = lane & 15;
  size_t tm = (size_t)wave * 16;
  v16bf a = load_a_f32(rel + (tm + nn) * RB_, 0, half);
  v16bf b = load_b_bf16(rpw + (size_t)nn * RB_, 0, half);
  v8f acc = {};
  acc = wmma_bf16(a, b, acc);
  if (nn < NH_) {
#pragma unroll
    for (int r = 0; r < 8; ++r)
      bias[(size_t)nn * S_ * S_ + tm + r + 8 * half] = (__bf16)acc[r];
  }
}

// -------------------------------------------------- unary: h @ key_unary_w^T -> [B,NH,S] f32
__global__ void k_unary(const __bf16* __restrict__ hb, const __bf16* __restrict__ kuw,
                        float* __restrict__ un) {
  int wave = blockIdx.x * 8 + (threadIdx.x >> 5);
  int lane = threadIdx.x & 31;
  int half = lane >> 4, nn = lane & 15;
  int tm = wave * 16;
  const __bf16* arow = hb + (size_t)(tm + nn) * H_;
  const __bf16* wrow = kuw + (size_t)nn * H_;
  v8f acc = {};
#pragma unroll 4
  for (int k0 = 0; k0 < H_; k0 += 32)
    acc = wmma_bf16(load_a_bf16(arow, k0, half), load_b_bf16(wrow, k0, half), acc);
  if (nn < NH_) {
#pragma unroll
    for (int r = 0; r < 8; ++r) {
      int m = tm + r + 8 * half;
      int b = m >> 11, s = m & (S_ - 1);
      un[(size_t)(b * NH_ + nn) * S_ + s] = acc[r];
    }
  }
}

// -------------------------------------------------- scores = q@k^T + bias + unary, mask
__global__ void k_scores(const __bf16* __restrict__ qb, const __bf16* __restrict__ kb,
                         const __bf16* __restrict__ bias, const float* __restrict__ un,
                         const unsigned char* __restrict__ mask, __bf16* __restrict__ sc) {
  int wave = blockIdx.x * 8 + (threadIdx.x >> 5);
  int lane = threadIdx.x & 31;
  int half = lane >> 4, nn = lane & 15;
  int kt = wave & 127;
  int qt = (wave >> 7) & 127;
  int bh = wave >> 14;            // b*NH + head, 0..23
  int head = bh % NH_, b = bh / NH_;
  const __bf16* qrow = qb + ((size_t)bh * S_ + qt * 16 + nn) * DH_;
  const __bf16* krow = kb + ((size_t)bh * S_ + kt * 16 + nn) * DH_;
  v8f acc = {};
  acc = wmma_bf16(load_a_bf16(qrow, 0, half), load_b_bf16(krow, 0, half), acc);
  acc = wmma_bf16(load_a_bf16(qrow, 32, half), load_b_bf16(krow, 32, half), acc);
  int key = kt * 16 + nn;
  float u = un[(size_t)bh * S_ + key];
  bool msk = mask[(size_t)b * S_ + key] != 0;
  const __bf16* brow = bias + (size_t)head * S_ * S_ + (size_t)(qt * 16) * S_ + key;
  __bf16* srow = sc + (size_t)bh * S_ * S_ + (size_t)(qt * 16) * S_ + key;
#pragma unroll
  for (int r = 0; r < 8; ++r) {
    int qr = r + 8 * half;
    float v = acc[r] + (float)brow[(size_t)qr * S_] + u;
    if (msk) v = -__builtin_inff();
    srow[(size_t)qr * S_] = (__bf16)v;
  }
}

// -------------------------------------------------- row softmax over 2048 keys (in place)
__global__ void k_softmax(__bf16* __restrict__ sc) {
  size_t row = blockIdx.x;
  __bf16* p = sc + row * S_;
  int tid = threadIdx.x, wid = tid >> 5, lane = tid & 31;
  v8bf v = *((const v8bf*)p + tid);
  float f[8];
  float mx = -__builtin_inff();
#pragma unroll
  for (int i = 0; i < 8; ++i) { f[i] = (float)v[i]; mx = fmaxf(mx, f[i]); }
  __shared__ float sm[8], ss[8];
  mx = wave_reduce_max(mx);
  if (lane == 0) sm[wid] = mx;
  __syncthreads();
  float bmax = sm[0];
#pragma unroll
  for (int w = 1; w < 8; ++w) bmax = fmaxf(bmax, sm[w]);
  float sum = 0.f;
#pragma unroll
  for (int i = 0; i < 8; ++i) { f[i] = __expf(f[i] - bmax); sum += f[i]; }
  sum = wave_reduce_sum(sum);
  if (lane == 0) ss[wid] = sum;
  __syncthreads();
  float tot = 0.f;
#pragma unroll
  for (int w = 0; w < 8; ++w) tot += ss[w];
  float inv = 1.0f / tot;
  v8bf o;
#pragma unroll
  for (int i = 0; i < 8; ++i) o[i] = (__bf16)(f[i] * inv);
  *((v8bf*)p + tid) = o;
}

// -------------------------------------------------- ctx = probs @ v   -> [B,S,H] f32
__global__ void k_ctx(const __bf16* __restrict__ pr, const __bf16* __restrict__ vb,
                      float* __restrict__ ctx) {
  int wave = blockIdx.x * 8 + (threadIdx.x >> 5);
  int lane = threadIdx.x & 31;
  int half = lane >> 4, nn = lane & 15;
  int dt = wave & 3;
  int qt = (wave >> 2) & 127;
  int bh = wave >> 9;
  int head = bh % NH_, b = bh / NH_;
  const __bf16* prow = pr + ((size_t)bh * S_ + qt * 16 + nn) * S_;
  const __bf16* vrow = vb + ((size_t)bh * DH_ + dt * 16 + nn) * S_;
  v8f acc = {};
#pragma unroll 4
  for (int k0 = 0; k0 < S_; k0 += 32)
    acc = wmma_bf16(load_a_bf16(prow, k0, half), load_b_bf16(vrow, k0, half), acc);
  int d = dt * 16 + nn;
#pragma unroll
  for (int r = 0; r < 8; ++r) {
    int q = qt * 16 + r + 8 * half;
    ctx[((size_t)b * S_ + q) * H_ + head * DH_ + d] = acc[r];
  }
}

// -------------------------------------------------- attn dense: ctx@Wo^T + bo + h -> f32
__global__ void k_dense_attn(const float* __restrict__ ctx, const __bf16* __restrict__ Wo,
                             const float* __restrict__ bo, const float* __restrict__ h,
                             float* __restrict__ pre) {
  int wave = blockIdx.x * 8 + (threadIdx.x >> 5);
  int lane = threadIdx.x & 31;
  int half = lane >> 4, nn = lane & 15;
  const int NT = H_ / 16;
  int tm = (wave / NT) * 16, tn = (wave % NT) * 16;
  const float*  arow = ctx + (size_t)(tm + nn) * H_;
  const __bf16* wrow = Wo + (size_t)(tn + nn) * H_;
  v8f acc = {};
  for (int k0 = 0; k0 < H_; k0 += 32)
    acc = wmma_bf16(load_a_f32(arow, k0, half), load_b_bf16(wrow, k0, half), acc);
  int n = tn + nn;
  float bb = bo[n];
#pragma unroll
  for (int r = 0; r < 8; ++r) {
    size_t idx = (size_t)(tm + r + 8 * half) * H_ + n;
    pre[idx] = acc[r] + bb + h[idx];
  }
}

// -------------------------------------------------- LayerNorm over H=768 (3 elems/thread)
__global__ void k_ln(const float* __restrict__ x, const float* __restrict__ g,
                     const float* __restrict__ bt, float* __restrict__ yf,
                     __bf16* __restrict__ yb) {
  size_t row = blockIdx.x;
  const float* p = x + row * H_;
  int tid = threadIdx.x, wid = tid >> 5, lane = tid & 31;
  float v0 = p[tid], v1 = p[tid + 256], v2 = p[tid + 512];
  __shared__ float s1[8], s2[8];
  float s = wave_reduce_sum(v0 + v1 + v2);
  if (lane == 0) s1[wid] = s;
  __syncthreads();
  float tot = 0.f;
#pragma unroll
  for (int w = 0; w < 8; ++w) tot += s1[w];
  float mean = tot * (1.0f / H_);
  float d0 = v0 - mean, d1 = v1 - mean, d2 = v2 - mean;
  float sq = wave_reduce_sum(d0 * d0 + d1 * d1 + d2 * d2);
  if (lane == 0) s2[wid] = sq;
  __syncthreads();
  float vtot = 0.f;
#pragma unroll
  for (int w = 0; w < 8; ++w) vtot += s2[w];
  float inv = rsqrtf(vtot * (1.0f / H_) + 1e-5f);
  float y0 = d0 * inv * g[tid] + bt[tid];
  float y1 = d1 * inv * g[tid + 256] + bt[tid + 256];
  float y2 = d2 * inv * g[tid + 512] + bt[tid + 512];
  float* yo = yf + row * H_;
  yo[tid] = y0; yo[tid + 256] = y1; yo[tid + 512] = y2;
  if (yb) {
    __bf16* o = yb + row * H_;
    o[tid] = (__bf16)y0; o[tid + 256] = (__bf16)y1; o[tid + 512] = (__bf16)y2;
  }
}

// -------------------------------------------------- FFN1 with TDM-staged weight tile
// Block = (n-tile, m-group of 8 tiles). The 16x768 bf16 weight tile (24KB) is DMA'd
// into LDS once by the Tensor Data Mover; all 8 waves stream B fragments from LDS.
__global__ void k_ffn1(const __bf16* __restrict__ ab, const __bf16* __restrict__ Wi,
                       const float* __restrict__ bi, __bf16* __restrict__ ib) {
  __shared__ __align__(16) __bf16 wtile[16 * H_];   // single static LDS alloc -> offset 0
  int tid = threadIdx.x;
  int wid = tid >> 5, lane = tid & 31;
  int half = lane >> 4, nn = lane & 15;
  const int NT = FF_ / 16;                 // 192 n-tiles
  int tnt = blockIdx.x % NT;               // n-tile (block-uniform)
  int mg  = blockIdx.x / NT;               // m-group (32 groups of 8 tiles)
  int tm  = (mg * 8 + wid) * 16;
  int tn  = tnt * 16;

#if __has_builtin(__builtin_amdgcn_tensor_load_to_lds)
  if (wid == 0) {
    unsigned long long ga =
        (unsigned long long)(uintptr_t)(Wi + (size_t)tn * H_);  // tile start
    // D# group0: count=1 | lds_addr | global_addr | type=2 ("image")
    u32x4 g0 = { 1u, 0u, (unsigned)(ga & 0xffffffffu),
                 (unsigned)(((ga >> 32) & 0x01ffffffu) | (2u << 30)) };
    // D# group1: data_size=2B; tensor 768x3072, tile 768x16, stride0=768
    i32x8 g1 = { (int)(1u << 16),            // data_size=1 (2 bytes)
                 (int)((H_  & 0xffff) << 16),// tensor_dim0 lo16 (bits 63:48)
                 (int)((FF_ & 0xffff) << 16),// dim0 hi16=0 | tensor_dim1 lo16
                 (int)((H_  & 0xffff) << 16),// dim1 hi16=0 | tile_dim0=768
                 16,                         // tile_dim1=16 | tile_dim2=0
                 H_,                         // tensor_dim0_stride lo32 = 768
                 0, 0 };
    i32x4 g2 = { 0, 0, 0, 0 };
    i32x4 g3 = { 0, 0, 0, 0 };
    i32x8 g4 = { 0, 0, 0, 0, 0, 0, 0, 0 };
    __builtin_amdgcn_tensor_load_to_lds(g0, g1, g2, g3, g4, 0);
    __builtin_amdgcn_s_wait_tensorcnt(0);
  }
#else
  for (int i = tid; i < 16 * H_ / 8; i += 256)
    ((v8bf*)wtile)[i] = ((const v8bf*)(Wi + (size_t)tn * H_))[i];
#endif
  // The TDM write is invisible to the compiler (it only sees descriptor ints).
  // Escape the LDS pointer *in addrspace(3)* through an asm with a memory
  // clobber: reads below can't fold to undef / move above the DMA, and they
  // keep their LDS provenance so they lower to ds_load_b128 (not flat).
  lds_bf16* wlds = (lds_bf16*)wtile;
  asm volatile("" : "+v"(wlds) : : "memory");
  __syncthreads();

  const __bf16* arow = ab + (size_t)(tm + nn) * H_;
  const lds_bf16* wrow = wlds + (size_t)nn * H_;   // LDS reads -> ds_load_b128
  v8f acc = {};
#pragma unroll 4
  for (int k0 = 0; k0 < H_; k0 += 32)
    acc = wmma_bf16(load_a_bf16(arow, k0, half), load_b_lds(wrow, k0, half), acc);
  int n = tn + nn;
  float bb = bi[n];
#pragma unroll
  for (int r = 0; r < 8; ++r) {
    float xv = acc[r] + bb;
    float gl = 0.5f * xv * (1.0f + erff(xv * 0.70710678118654752f));  // exact erf gelu
    ib[(size_t)(tm + r + 8 * half) * FF_ + n] = (__bf16)gl;
  }
}

// -------------------------------------------------- FFN2: inter@Wout^T + bout + attn -> f32
__global__ void k_ffn2(const __bf16* __restrict__ ib, const __bf16* __restrict__ Wout,
                       const float* __restrict__ bout, const float* __restrict__ res,
                       float* __restrict__ pre) {
  int wave = blockIdx.x * 8 + (threadIdx.x >> 5);
  int lane = threadIdx.x & 31;
  int half = lane >> 4, nn = lane & 15;
  const int NT = H_ / 16;
  int tm = (wave / NT) * 16, tn = (wave % NT) * 16;
  const __bf16* arow = ib + (size_t)(tm + nn) * FF_;
  const __bf16* wrow = Wout + (size_t)(tn + nn) * FF_;
  v8f acc = {};
  for (int k0 = 0; k0 < FF_; k0 += 32) {
    if (k0 + 96 < FF_) {  // uniform branch; exercises global_prefetch_b8
      __builtin_prefetch((const void*)(arow + k0 + 96), 0, 1);
      __builtin_prefetch((const void*)(wrow + k0 + 96), 0, 1);
    }
    acc = wmma_bf16(load_a_bf16(arow, k0, half), load_b_bf16(wrow, k0, half), acc);
  }
  int n = tn + nn;
  float bb = bout[n];
#pragma unroll
  for (int r = 0; r < 8; ++r) {
    size_t idx = (size_t)(tm + r + 8 * half) * H_ + n;
    pre[idx] = acc[r] + bb + res[idx];
  }
}

// ================================================================ launcher
extern "C" void kernel_launch(void* const* d_in, const int* in_sizes, int n_in,
                              void* d_out, int out_size, void* d_ws, size_t ws_size,
                              hipStream_t stream) {
  (void)in_sizes; (void)n_in; (void)out_size; (void)ws_size;
  const float* h    = (const float*)d_in[0];
  const unsigned char* mask = (const unsigned char*)d_in[1];
  const float* rel  = (const float*)d_in[2];
  const float* Wq   = (const float*)d_in[3];
  const float* bq   = (const float*)d_in[4];
  const float* Wk   = (const float*)d_in[5];
  const float* Wv   = (const float*)d_in[6];
  const float* bv   = (const float*)d_in[7];
  const float* rpw  = (const float*)d_in[8];
  const float* kuw  = (const float*)d_in[9];
  const float* Wo   = (const float*)d_in[10];
  const float* bo   = (const float*)d_in[11];
  const float* ln1g = (const float*)d_in[12];
  const float* ln1b = (const float*)d_in[13];
  const float* Wi   = (const float*)d_in[14];
  const float* bi   = (const float*)d_in[15];
  const float* Wout = (const float*)d_in[16];
  const float* bout = (const float*)d_in[17];
  const float* ln2g = (const float*)d_in[18];
  const float* ln2b = (const float*)d_in[19];
  float* out = (float*)d_out;

  char* ws = (char*)d_ws;
  size_t off = 0;
  auto alloc = [&](size_t bytes) -> void* {
    void* p = ws + off;
    off = (off + bytes + 255) & ~(size_t)255;
    return p;
  };
  __bf16* hb    = (__bf16*)alloc((size_t)MTOK * H_ * 2);
  __bf16* Wqb   = (__bf16*)alloc((size_t)H_ * H_ * 2);
  __bf16* Wkb   = (__bf16*)alloc((size_t)H_ * H_ * 2);
  __bf16* Wvb   = (__bf16*)alloc((size_t)H_ * H_ * 2);
  __bf16* Wob   = (__bf16*)alloc((size_t)H_ * H_ * 2);
  __bf16* Wib   = (__bf16*)alloc((size_t)FF_ * H_ * 2);
  __bf16* Woutb = (__bf16*)alloc((size_t)H_ * FF_ * 2);
  __bf16* kuwb  = (__bf16*)alloc((size_t)16 * H_ * 2);
  __bf16* rpwb  = (__bf16*)alloc((size_t)16 * RB_ * 2);
  __bf16* qb    = (__bf16*)alloc((size_t)MTOK * H_ * 2);
  __bf16* kb    = (__bf16*)alloc((size_t)MTOK * H_ * 2);
  __bf16* vb    = (__bf16*)alloc((size_t)MTOK * H_ * 2);
  float*  un    = (float*)alloc((size_t)B_ * NH_ * S_ * 4);
  __bf16* bias  = (__bf16*)alloc((size_t)NH_ * S_ * S_ * 2);
  __bf16* sc    = (__bf16*)alloc((size_t)B_ * NH_ * S_ * S_ * 2);
  float*  ctx   = (float*)alloc((size_t)MTOK * H_ * 4);
  float*  pre1  = (float*)alloc((size_t)MTOK * H_ * 4);
  float*  attnf = (float*)alloc((size_t)MTOK * H_ * 4);
  __bf16* attnb = (__bf16*)alloc((size_t)MTOK * H_ * 2);
  __bf16* ib    = (__bf16*)alloc((size_t)MTOK * FF_ * 2);
  float*  pre2  = (float*)alloc((size_t)MTOK * H_ * 4);

  auto cvt = [&](const float* s, __bf16* d, int n) {
    k_cvt<<<dim3((n + 255) / 256), dim3(256), 0, stream>>>(s, d, n);
  };
  cvt(h, hb, MTOK * H_);
  cvt(Wq, Wqb, H_ * H_);  cvt(Wk, Wkb, H_ * H_);
  cvt(Wv, Wvb, H_ * H_);  cvt(Wo, Wob, H_ * H_);
  cvt(Wi, Wib, FF_ * H_); cvt(Wout, Woutb, H_ * FF_);
  k_cvt_pad<<<dim3((16 * H_ + 255) / 256), dim3(256), 0, stream>>>(kuw, kuwb, NH_, H_, 16);
  k_cvt_pad<<<dim3((16 * RB_ + 255) / 256), dim3(256), 0, stream>>>(rpw, rpwb, NH_, RB_, 16);

  k_qkv<<<dim3(3 * (MTOK / 16) * (H_ / 16) / 8), dim3(256), 0, stream>>>(
      hb, Wqb, Wkb, Wvb, bq, bv, qb, kb, vb);
  k_relpos<<<dim3((S_ * S_ / 16) / 8), dim3(256), 0, stream>>>(rel, rpwb, bias);
  k_unary<<<dim3((MTOK / 16) / 8), dim3(256), 0, stream>>>(hb, kuwb, un);
  k_scores<<<dim3(B_ * NH_ * 128 * 128 / 8), dim3(256), 0, stream>>>(
      qb, kb, bias, un, mask, sc);
  k_softmax<<<dim3(B_ * NH_ * S_), dim3(256), 0, stream>>>(sc);
  k_ctx<<<dim3(B_ * NH_ * 128 * 4 / 8), dim3(256), 0, stream>>>(sc, vb, ctx);
  k_dense_attn<<<dim3((MTOK / 16) * (H_ / 16) / 8), dim3(256), 0, stream>>>(
      ctx, Wob, bo, h, pre1);
  k_ln<<<dim3(MTOK), dim3(256), 0, stream>>>(pre1, ln1g, ln1b, attnf, attnb);
  k_ffn1<<<dim3((FF_ / 16) * (MTOK / 16 / 8)), dim3(256), 0, stream>>>(attnb, Wib, bi, ib);
  k_ffn2<<<dim3((MTOK / 16) * (H_ / 16) / 8), dim3(256), 0, stream>>>(
      ib, Woutb, bout, attnf, pre2);
  k_ln<<<dim3(MTOK), dim3(256), 0, stream>>>(pre2, ln2g, ln2b, out, (__bf16*)nullptr);
}
